// Latent_SSM_Irregular_Mesh_23845658428254
// MI455X (gfx1250) — matrix operand order repaired and checked by reading, not compiled, not run on hardware
//
#include <hip/hip_runtime.h>
#include <hip/hip_bf16.h>

typedef __attribute__((ext_vector_type(16))) _Float16 v16h;
typedef __attribute__((ext_vector_type(8)))  float    v8f;
typedef __attribute__((ext_vector_type(4)))  unsigned int u32x4;
typedef __attribute__((ext_vector_type(4)))  int i32x4;
typedef __attribute__((ext_vector_type(8)))  int i32x8;

#define DEV __device__ __forceinline__

#if defined(__HIP_DEVICE_COMPILE__) && __has_builtin(__builtin_amdgcn_tensor_load_to_lds)
#define HAS_TDM 1
#else
#define HAS_TDM 0
#endif

DEV int lane_id() { return (int)(threadIdx.x & 31); }

// ---------------- WMMA helpers (CDNA5 16x16x32 f16 -> f32) ----------------
DEV v8f wmma32(v16h a, v16h b, v8f c) {
  return __builtin_amdgcn_wmma_f32_16x16x32_f16(false, a, false, b, (short)0, c,
                                                false, false);
}

// A tile 16x32, logical A[m][k] = src[(row0+m)*lda + k0+k], fp32 source
DEV v16h load_a_rm_f32(const float* src, long lda, int row0, int k0) {
  int l = lane_id();
  int m = l & 15;
  int kb = (l & 16) ? 8 : 0;
  const float* p = src + (long)(row0 + m) * lda + k0 + kb;
  v16h a;
#pragma unroll
  for (int e = 0; e < 8; ++e) a[e] = (_Float16)p[e];
#pragma unroll
  for (int e = 0; e < 8; ++e) a[8 + e] = (_Float16)p[16 + e];
  return a;
}

// A tile 16x32, f16 source, row-major
DEV v16h load_a_rm_f16(const _Float16* src, long lda, int row0, int k0) {
  int l = lane_id();
  int m = l & 15;
  int kb = (l & 16) ? 8 : 0;
  const _Float16* p = src + (long)(row0 + m) * lda + k0 + kb;
  v16h a;
#pragma unroll
  for (int e = 0; e < 8; ++e) a[e] = p[e];
#pragma unroll
  for (int e = 0; e < 8; ++e) a[8 + e] = p[16 + e];
  return a;
}

// A tile 16x32 from transposed storage: logical A[m][k] = src[(k0+k)*lda + row0+m]
DEV v16h load_a_cm_f16(const _Float16* src, long lda, int row0, int k0) {
  int l = lane_id();
  int m = l & 15;
  int kb = (l & 16) ? 8 : 0;
  const _Float16* p = src + (long)(k0 + kb) * lda + row0 + m;
  v16h a;
#pragma unroll
  for (int e = 0; e < 8; ++e) a[e] = p[(long)e * lda];
#pragma unroll
  for (int e = 0; e < 8; ++e) a[8 + e] = p[(long)(16 + e) * lda];
  return a;
}

// B tile 32x16, logical B[k][n] = src[(k0+k)*ldb + n0+n], f16 source row-major
DEV v16h load_b_rm_f16(const _Float16* src, long ldb, int k0, int n0) {
  int l = lane_id();
  int n = n0 + (l & 15);
  int kb = k0 + ((l & 16) ? 8 : 0);
  const _Float16* p = src + (long)kb * ldb + n;
  v16h b;
#pragma unroll
  for (int e = 0; e < 8; ++e) b[e] = p[(long)e * ldb];
#pragma unroll
  for (int e = 0; e < 8; ++e) b[8 + e] = p[(long)(16 + e) * ldb];
  return b;
}

// B tile 32x16 with logical B[k][n] = W[n][k]  (W row-major (N,K), fp32)
DEV v16h load_bt_f32(const float* W, long ldb, int k0, int n0) {
  int l = lane_id();
  int n = n0 + (l & 15);
  int kb = k0 + ((l & 16) ? 8 : 0);
  const float* p = W + (long)n * ldb + kb;
  v16h b;
#pragma unroll
  for (int e = 0; e < 8; ++e) b[e] = (_Float16)p[e];
#pragma unroll
  for (int e = 0; e < 8; ++e) b[8 + e] = (_Float16)p[16 + e];
  return b;
}

DEV void store_d_f32(float* dst, long ldd, int row0, int col0, v8f d) {
  int l = lane_id();
  int n = l & 15;
  int mb = (l & 16) ? 8 : 0;
#pragma unroll
  for (int v = 0; v < 8; ++v)
    dst[(long)(row0 + mb + v) * ldd + col0 + n] = d[v];
}

// ---------------- TDM: 2D tile (fp32) global -> LDS -------------------------
#if HAS_TDM
DEV unsigned lds_off(const void* p) {  // low 32 bits of flat addr == LDS offset
  return (unsigned)(uintptr_t)p;
}
DEV void tdm_load_2d(unsigned ldsa, const float* gaddr, unsigned tx,
                     unsigned ty, unsigned stride_elems) {
  unsigned long long ga = (unsigned long long)(uintptr_t)gaddr;
  u32x4 g0;
  g0[0] = 1u;                                   // count=1 (user descriptor)
  g0[1] = ldsa;                                 // lds_addr
  g0[2] = (unsigned)(ga & 0xffffffffu);         // global_addr[31:0]
  g0[3] = (unsigned)((ga >> 32) & 0x1ffffffu) | (2u << 30);  // addr hi | type=2
  i32x8 g1;
  g1[0] = (int)(2u << 16);                      // workgroup_mask=0, data_size=4B
  g1[1] = (int)((tx & 0xffffu) << 16);          // tensor_dim0 lo
  g1[2] = (int)(((tx >> 16) & 0xffffu) | ((ty & 0xffffu) << 16));  // d0 hi|d1 lo
  g1[3] = (int)(((ty >> 16) & 0xffffu) | ((tx & 0xffffu) << 16));  // d1 hi|tile0
  g1[4] = (int)(ty & 0xffffu);                  // tile_dim1, tile_dim2=0
  g1[5] = (int)stride_elems;                    // tensor_dim0_stride lo
  g1[6] = 0;                                    // stride0 hi | stride1 lo
  g1[7] = 0;                                    // stride1 hi
  i32x4 z4 = {0, 0, 0, 0};
#if __clang_major__ >= 23
  i32x8 z8 = {0, 0, 0, 0, 0, 0, 0, 0};
  __builtin_amdgcn_tensor_load_to_lds(g0, g1, z4, z4, z8, 0);
#else
  __builtin_amdgcn_tensor_load_to_lds(g0, g1, z4, z4, 0);
#endif
}
#endif

// ------- Kernel 1/2/9: out = f16( A(f32) @ W^T + bias ), 4 N-tiles/wave -----
__global__ void k_gemm_xw_f16(const float* __restrict__ A,
                              const float* __restrict__ W,
                              const float* __restrict__ bias,
                              _Float16* __restrict__ out, int M, int K, int N) {
  int wave = blockIdx.x * (blockDim.x >> 5) + (threadIdx.x >> 5);
  int ngrp = N >> 6;  // groups of 4 n-tiles
  int tm = wave / ngrp, tg = wave % ngrp;
  if (tm * 16 >= M) return;  // wave-uniform
  v8f acc0 = {}, acc1 = {}, acc2 = {}, acc3 = {};
  for (int k = 0; k < K; k += 32) {
    v16h a = load_a_rm_f32(A, K, tm * 16, k);
    if (k + 32 < K)
      __builtin_prefetch(A + (long)(tm * 16 + (lane_id() & 15)) * K + k + 32, 0, 0);
    v16h b0 = load_bt_f32(W, K, k, tg * 64);
    v16h b1 = load_bt_f32(W, K, k, tg * 64 + 16);
    v16h b2 = load_bt_f32(W, K, k, tg * 64 + 32);
    v16h b3 = load_bt_f32(W, K, k, tg * 64 + 48);
    acc0 = wmma32(a, b0, acc0);
    acc1 = wmma32(a, b1, acc1);
    acc2 = wmma32(a, b2, acc2);
    acc3 = wmma32(a, b3, acc3);
  }
  int l = lane_id();
  int n16 = l & 15;
  int mb = (l & 16) ? 8 : 0;
#pragma unroll
  for (int j = 0; j < 4; ++j) {
    v8f acc = (j == 0) ? acc0 : (j == 1) ? acc1 : (j == 2) ? acc2 : acc3;
    int n = tg * 64 + j * 16 + n16;
    float bv = bias ? bias[n] : 0.f;
#pragma unroll
    for (int v = 0; v < 8; ++v)
      out[(long)(tm * 16 + mb + v) * N + n] = (_Float16)(acc[v] + bv);
  }
}

// ---------------- Kernel 5: GEMM f32 out (zxbcdt = st @ in_w^T) -------------
__global__ void k_gemm_f32out(const float* __restrict__ A,
                              const float* __restrict__ W,
                              float* __restrict__ out, int M, int K, int N) {
  int wave = blockIdx.x * (blockDim.x >> 5) + (threadIdx.x >> 5);
  int ntiles = N >> 4;
  int tm = wave / ntiles, tn = wave % ntiles;
  if (tm * 16 >= M) return;
  v8f acc = {};
  for (int k = 0; k < K; k += 32) {
    v16h a = load_a_rm_f32(A, K, tm * 16, k);
    v16h b = load_bt_f32(W, K, k, tn * 16);
    acc = wmma32(a, b, acc);
  }
  store_d_f32(out, N, tm * 16, tn * 16, acc);
}

// ---------------- Kernel 3: slice logits + softmax --------------------------
__global__ void k_slice_softmax(const _Float16* __restrict__ xm,
                                const float* __restrict__ Wslice,
                                const float* __restrict__ bslice,
                                const float* __restrict__ temp,
                                _Float16* __restrict__ sw, int Bsz, int N) {
  __shared__ float WsT[64 * 64];  // WsT[p*64+g] = Wslice[g*64+p]
  for (int i = threadIdx.x; i < 64 * 64; i += blockDim.x) {
    int g = i >> 6, p = i & 63;
    WsT[p * 64 + g] = Wslice[i];
  }
  __syncthreads();
  long t = (long)blockIdx.x * blockDim.x + threadIdx.x;
  long total = (long)Bsz * N * 8;
  if (t >= total) return;
  int h = (int)(t % 8);
  long bn = t / 8;
  int n = (int)(bn % N);
  int b = (int)(bn / N);
  const _Float16* xr = xm + bn * 512 + h * 64;
  float acc[64];
#pragma unroll
  for (int g = 0; g < 64; ++g) acc[g] = bslice[g];
  for (int p = 0; p < 64; ++p) {
    float xv = (float)xr[p];
#pragma unroll
    for (int g = 0; g < 64; ++g) acc[g] += xv * WsT[p * 64 + g];
  }
  float tinv = 1.f / fmaxf(temp[h], 1e-6f);
  float mx = -1e30f;
#pragma unroll
  for (int g = 0; g < 64; ++g) { acc[g] *= tinv; mx = fmaxf(mx, acc[g]); }
  float s = 0.f;
#pragma unroll
  for (int g = 0; g < 64; ++g) { acc[g] = __expf(acc[g] - mx); s += acc[g]; }
  float inv = 1.f / s;
  _Float16* o = sw + (((long)(b * 8 + h) * N) + n) * 64;
#pragma unroll
  for (int g = 0; g < 64; ++g) o[g] = (_Float16)(acc[g] * inv);
}

// ---------------- Kernel 4: st = (sw^T @ fx) / (snorm + 1e-5) ---------------
__global__ void k_st_gemm(const _Float16* __restrict__ sw,
                          const _Float16* __restrict__ fx,
                          float* __restrict__ st, int N) {
  int b = blockIdx.x >> 3, h = blockIdx.x & 7;
  const _Float16* swp = sw + (long)(b * 8 + h) * N * 64;
  __shared__ float snorm[64];
  __shared__ float partial[8][64];
  int t = threadIdx.x, g = t & 63, ch = t >> 6;
  float s = 0.f;
  for (int n = ch; n < N; n += 8) s += (float)swp[(long)n * 64 + g];
  partial[ch][g] = s;
  __syncthreads();
  if (t < 64) {
    float tot = 0.f;
#pragma unroll
    for (int c = 0; c < 8; ++c) tot += partial[c][t];
    snorm[t] = 1.f / (tot + 1e-5f);
  }
  __syncthreads();
  int wave = t >> 5;
  int tg = wave >> 2, tc = wave & 3;
  const _Float16* Bp = fx + (long)b * N * 512 + h * 64;
  v8f acc = {};
  for (int k = 0; k < N; k += 32) {
    v16h a = load_a_cm_f16(swp, 64, tg * 16, k);   // A[g][n] = sw[n][g]
    v16h bb = load_b_rm_f16(Bp, 512, k, tc * 16);  // B[n][c]
    acc = wmma32(a, bb, acc);
  }
  int l = lane_id();
  int n16 = l & 15;
  int mb = (l & 16) ? 8 : 0;
  float* o = st + (long)b * 64 * 512 + h * 64;
#pragma unroll
  for (int v = 0; v < 8; ++v) {
    int gg = tg * 16 + mb + v;
    o[(long)gg * 512 + tc * 16 + n16] = acc[v] * snorm[gg];
  }
}

// ---------------- Kernel 6: depthwise causal conv (K=3) + SiLU --------------
__global__ void k_conv_silu(const float* __restrict__ zx,
                            const float* __restrict__ conv_w,
                            const float* __restrict__ conv_b,
                            float* __restrict__ xbc_act) {
  int idx = blockIdx.x * blockDim.x + threadIdx.x;
  if (idx >= 4 * 64 * 640) return;
  int c = idx % 640;
  int bl = idx / 640;
  int l = bl % 64, b = bl / 64;
  const float* col = zx + 512 + c;
  float acc = conv_b[c];
#pragma unroll
  for (int k = 0; k < 3; ++k) {
    int ls = l + k - 2;
    if (ls >= 0) acc += col[(long)(b * 64 + ls) * 1168] * conv_w[c * 3 + k];
  }
  float sig = 1.f / (1.f + __expf(-acc));
  xbc_act[((long)b * 64 + l) * 640 + c] = acc * sig;
}

// ---------------- Kernel 7: bidirectional SSM scan (TDM-staged LDS) ---------
__global__ void k_scan(const float* __restrict__ zx,
                       const float* __restrict__ xbc_act,
                       const float* __restrict__ dt_bias,
                       const float* __restrict__ A_log,
                       float* __restrict__ yscan) {
  int bb = blockIdx.x >> 3, h = blockIdx.x & 7;
  int b = bb & 3;
  bool rev = bb >= 4;
  __shared__ float Xs[64][64];  // forward (unflipped) order
  __shared__ float Bs[64][64];
  __shared__ float Cs[64][64];
  __shared__ float dAs[64], dts[64];
  __shared__ float red[4][64];
  int t = threadIdx.x;
  const float* base = xbc_act + (long)b * 64 * 640;
#if HAS_TDM
  if (t < 32) {  // wave 0 drives the Tensor Data Mover
    tdm_load_2d(lds_off(&Xs[0][0]), base + h * 64, 64, 64, 640);
    tdm_load_2d(lds_off(&Bs[0][0]), base + 512, 64, 64, 640);
    tdm_load_2d(lds_off(&Cs[0][0]), base + 576, 64, 64, 640);
    __builtin_amdgcn_s_wait_tensorcnt(0);
  }
#else
  for (int i = t; i < 4096; i += 256) {
    int r = i >> 6, j = i & 63;
    const float* row = base + (long)r * 640;
    Xs[r][j] = row[h * 64 + j];
    Bs[r][j] = row[512 + j];
    Cs[r][j] = row[576 + j];
  }
#endif
  if (t < 64) {
    int l = t;
    int ls = rev ? 63 - l : l;
    int hcol = rev ? (8 + h) : h;
    float dtr = zx[((long)b * 64 + ls) * 1168 + 1152 + hcol] + dt_bias[h];
    float dt = (dtr > 20.f) ? dtr : log1pf(__expf(dtr));  // softplus
    dts[l] = dt;
    dAs[l] = __expf(dt * -__expf(A_log[h]));
  }
  __syncthreads();
  int p = t & 63, nb = (t >> 6) * 16;
  float hst[16];
#pragma unroll
  for (int i = 0; i < 16; ++i) hst[i] = 0.f;
  for (int l = 0; l < 64; ++l) {
    int sl = rev ? 63 - l : l;  // flip at consumption
    float dA = dAs[l];
    float xv = Xs[sl][p] * dts[l];
    float part = 0.f;
#pragma unroll
    for (int i = 0; i < 16; ++i) {
      hst[i] = hst[i] * dA + Bs[sl][nb + i] * xv;
      part += hst[i] * Cs[sl][nb + i];
    }
    red[t >> 6][p] = part;
    __syncthreads();
    if (t < 64) {
      float y = red[0][t] + red[1][t] + red[2][t] + red[3][t];
      yscan[(((long)bb * 64 + l) * 8 + h) * 64 + t] = y;
    }
    __syncthreads();
  }
}

// ---------------- Kernel 8: shift-combine, D-residual, gate, RMSNorm --------
__global__ void k_post(const float* __restrict__ zx,
                       const float* __restrict__ xbc_act,
                       const float* __restrict__ yscan,
                       const float* __restrict__ fcD_w,
                       const float* __restrict__ Dp,
                       const float* __restrict__ norm_w,
                       float* __restrict__ yg_out) {
  int b = blockIdx.x >> 6, l = blockIdx.x & 63;
  __shared__ float xs[512];
  __shared__ float dv[8];
  __shared__ float red[256];
  __shared__ float rinv_s;
  int t = threadIdx.x;
  const float* xrow = xbc_act + ((long)b * 64 + l) * 640;
  xs[t] = xrow[t];
  xs[t + 256] = xrow[t + 256];
  __syncthreads();
  for (int h = 0; h < 8; ++h) {
    float pp = xs[t] * fcD_w[h * 512 + t] + xs[t + 256] * fcD_w[h * 512 + t + 256];
    red[t] = pp;
    __syncthreads();
    for (int s2 = 128; s2 > 0; s2 >>= 1) {
      if (t < s2) red[t] += red[t + s2];
      __syncthreads();
    }
    if (t == 0) dv[h] = red[0] + Dp[h];
    __syncthreads();
  }
  float yv[2];
#pragma unroll
  for (int i = 0; i < 2; ++i) {
    int c = t + i * 256;
    float yfw = (l > 0) ? yscan[((long)b * 64 + (l - 1)) * 512 + c] : 0.f;
    float ybw = (l < 63) ? yscan[((long)(b + 4) * 64 + (62 - l)) * 512 + c] : 0.f;
    float y = yfw + ybw;
    y += xs[c] * dv[c >> 6];
    float z = zx[((long)b * 64 + l) * 1168 + c];
    float sz = z / (1.f + __expf(-z));  // silu(z)
    yv[i] = y * sz;
  }
  red[t] = yv[0] * yv[0] + yv[1] * yv[1];
  __syncthreads();
  for (int s2 = 128; s2 > 0; s2 >>= 1) {
    if (t < s2) red[t] += red[t + s2];
    __syncthreads();
  }
  if (t == 0) rinv_s = rsqrtf(red[0] / 512.f + 1e-5f);
  __syncthreads();
  float* orow = yg_out + ((long)b * 64 + l) * 512;
#pragma unroll
  for (int i = 0; i < 2; ++i) {
    int c = t + i * 256;
    orow[c] = yv[i] * rinv_s * norm_w[c];
  }
}

// ------- Kernel 10: ymid = sw @ ot per (b,h); 4 N-tiles/wave, A reused ------
__global__ void k_unslice(const _Float16* __restrict__ sw,
                          const _Float16* __restrict__ ot16,
                          _Float16* __restrict__ ymid, int N) {
  int wave = blockIdx.x * (blockDim.x >> 5) + (threadIdx.x >> 5);
  int tm = wave & 511;
  int bh = wave >> 9;
  int b = bh >> 3, h = bh & 7;
  const _Float16* swp = sw + (long)bh * N * 64;
  const _Float16* Bp = ot16 + (long)b * 64 * 512 + h * 64;  // B[g][p]
  v8f acc0 = {}, acc1 = {}, acc2 = {}, acc3 = {};
#pragma unroll
  for (int k = 0; k < 64; k += 32) {
    v16h a = load_a_rm_f16(swp, 64, tm * 16, k);
    v16h b0 = load_b_rm_f16(Bp, 512, k, 0);
    v16h b1 = load_b_rm_f16(Bp, 512, k, 16);
    v16h b2 = load_b_rm_f16(Bp, 512, k, 32);
    v16h b3 = load_b_rm_f16(Bp, 512, k, 48);
    acc0 = wmma32(a, b0, acc0);
    acc1 = wmma32(a, b1, acc1);
    acc2 = wmma32(a, b2, acc2);
    acc3 = wmma32(a, b3, acc3);
  }
  int l = lane_id();
  int n16 = l & 15;
  int mb = (l & 16) ? 8 : 0;
#pragma unroll
  for (int j = 0; j < 4; ++j) {
    v8f acc = (j == 0) ? acc0 : (j == 1) ? acc1 : (j == 2) ? acc2 : acc3;
    _Float16* o = ymid + (long)b * N * 512 + h * 64 + j * 16 + n16;
#pragma unroll
    for (int v = 0; v < 8; ++v)
      o[(long)(tm * 16 + mb + v) * 512] = (_Float16)acc[v];
  }
}

// ------- Kernel 11: out = ymid @ out_w^T + out_b; 4 N-tiles/wave ------------
__global__ void k_final(const _Float16* __restrict__ ymid,
                        const float* __restrict__ out_w,
                        const float* __restrict__ out_b,
                        float* __restrict__ out, int M) {
  int wave = blockIdx.x * (blockDim.x >> 5) + (threadIdx.x >> 5);
  int tg = wave & 3;  // 4 groups of 4 tiles -> N=256
  int tm = wave >> 2;
  v8f acc0 = {}, acc1 = {}, acc2 = {}, acc3 = {};
  for (int k = 0; k < 512; k += 32) {
    v16h a = load_a_rm_f16(ymid, 512, tm * 16, k);
    v16h b0 = load_bt_f32(out_w, 512, k, tg * 64);
    v16h b1 = load_bt_f32(out_w, 512, k, tg * 64 + 16);
    v16h b2 = load_bt_f32(out_w, 512, k, tg * 64 + 32);
    v16h b3 = load_bt_f32(out_w, 512, k, tg * 64 + 48);
    acc0 = wmma32(a, b0, acc0);
    acc1 = wmma32(a, b1, acc1);
    acc2 = wmma32(a, b2, acc2);
    acc3 = wmma32(a, b3, acc3);
  }
  int l = lane_id();
  int n16 = l & 15;
  int mb = (l & 16) ? 8 : 0;
#pragma unroll
  for (int j = 0; j < 4; ++j) {
    v8f acc = (j == 0) ? acc0 : (j == 1) ? acc1 : (j == 2) ? acc2 : acc3;
    int n = tg * 64 + j * 16 + n16;
    float bv = out_b[n];
#pragma unroll
    for (int v = 0; v < 8; ++v)
      out[(long)(tm * 16 + mb + v) * 256 + n] = acc[v] + bv;
  }
}

// ---------------------------- launcher --------------------------------------
extern "C" void kernel_launch(void* const* d_in, const int* in_sizes, int n_in,
                              void* d_out, int out_size, void* d_ws,
                              size_t ws_size, hipStream_t stream) {
  const float* x        = (const float*)d_in[0];   // (4,8192,256)
  const float* W_fx     = (const float*)d_in[1];   // (512,256)
  const float* b_fx     = (const float*)d_in[2];
  const float* W_x      = (const float*)d_in[3];
  const float* b_x      = (const float*)d_in[4];
  const float* W_slice  = (const float*)d_in[5];   // (64,64)
  const float* b_slice  = (const float*)d_in[6];
  const float* temp     = (const float*)d_in[7];   // (8)
  const float* hy_in_w  = (const float*)d_in[8];   // (1168,512)
  const float* hy_cw    = (const float*)d_in[9];   // (640,3)
  const float* hy_cb    = (const float*)d_in[10];
  const float* dt_bias  = (const float*)d_in[11];
  const float* A_log    = (const float*)d_in[12];
  const float* hy_D     = (const float*)d_in[13];
  const float* fcD_w    = (const float*)d_in[14];  // (8,512)
  const float* norm_w   = (const float*)d_in[15];
  const float* hy_out_w = (const float*)d_in[16];  // (512,512)
  const float* out_w    = (const float*)d_in[17];  // (256,512)
  const float* out_b    = (const float*)d_in[18];
  float* out = (float*)d_out;

  const int Bsz = 4, N = 8192, M = Bsz * N;  // 32768 rows

  char* ws = (char*)d_ws;
  _Float16* fx16 = (_Float16*)ws;                    // 32 MiB
  _Float16* xm16 = (_Float16*)(ws + 33554432ull);    // 32 MiB (reused: ymid)
  _Float16* sw16 = (_Float16*)(ws + 67108864ull);    // 32 MiB
  char* sm = ws + 100663296ull;
  float* st    = (float*)(sm);                          // 256*512*4
  float* zx    = (float*)(sm + 524288ull);              // 256*1168*4
  float* xbc   = (float*)(sm + 524288ull + 1196032ull); // 4*64*640*4
  float* yscan = (float*)(sm + 524288ull + 1196032ull + 655360ull);
  float* yg    = (float*)(sm + 524288ull + 1196032ull + 655360ull + 1048576ull);
  _Float16* ot16 = (_Float16*)(sm + 524288ull + 1196032ull + 655360ull +
                               1048576ull + 524288ull);
  _Float16* ymid16 = xm16;

  // 1-2: fx, xm projections (WMMA, 4 tiles/wave): 2048*8 waves -> 2048 blocks
  k_gemm_xw_f16<<<2048, 256, 0, stream>>>(x, W_fx, b_fx, fx16, M, 256, 512);
  k_gemm_xw_f16<<<2048, 256, 0, stream>>>(x, W_x, b_x, xm16, M, 256, 512);
  // 3: slice softmax
  k_slice_softmax<<<1024, 256, 0, stream>>>(xm16, W_slice, b_slice, temp, sw16,
                                            Bsz, N);
  // 4: st = (sw^T @ fx)/snorm
  k_st_gemm<<<32, 512, 0, stream>>>(sw16, fx16, st, N);
  // 5: zxbcdt = st @ in_w^T: 16*73 = 1168 tiles -> 146 blocks
  k_gemm_f32out<<<146, 256, 0, stream>>>(st, hy_in_w, zx, 256, 512, 1168);
  // 6: conv + silu
  k_conv_silu<<<640, 256, 0, stream>>>(zx, hy_cw, hy_cb, xbc);
  // 7: bidirectional scan (TDM-staged)
  k_scan<<<64, 256, 0, stream>>>(zx, xbc, dt_bias, A_log, yscan);
  // 8: combine + gate + rmsnorm
  k_post<<<256, 256, 0, stream>>>(zx, xbc, yscan, fcD_w, hy_D, norm_w, yg);
  // 9: ot = yg @ hy_out_w^T: 16*8 = 128 waves -> 16 blocks
  k_gemm_xw_f16<<<16, 256, 0, stream>>>(yg, hy_out_w, nullptr, ot16, 256, 512,
                                        512);
  // 10: ymid = sw @ ot per (b,h): 32*512 waves -> 2048 blocks
  k_unslice<<<2048, 256, 0, stream>>>(sw16, ot16, ymid16, N);
  // 11: out = ymid @ out_w^T + b: 2048*4 waves -> 1024 blocks
  k_final<<<1024, 256, 0, stream>>>(ymid16, out_w, out_b, out, M);
}
